// GIN_DGL_58110907515583
// MI455X (gfx1250) — matrix operand order repaired and checked by reading, not compiled, not run on hardware
//
#include <hip/hip_runtime.h>
#include <hip/hip_bf16.h>
#include <stdint.h>

typedef __attribute__((ext_vector_type(16))) __bf16 v16bf;
typedef __attribute__((ext_vector_type(8)))  float  v8f;

#define N_NODES 100000
#define N_EDGES 1600000
#define IN_DIM  128
#define HID     96
#define NCLS    16
#define BN_EPS  1e-5f

// ---------- helpers ----------
__device__ __forceinline__ unsigned short f2bf(float f) {
    union { float f; uint32_t u; } c; c.f = f;
    uint32_t u = c.u;
    return (unsigned short)((u + 0x7fffu + ((u >> 16) & 1u)) >> 16);
}

__global__ void k_zero(float* __restrict__ p, int n) {
    int i = blockIdx.x * blockDim.x + threadIdx.x;
    int stride = gridDim.x * blockDim.x;
    for (; i < n; i += stride) p[i] = 0.f;
}

// ---------- A-fragment pack: [rowtile][ktile][lane 32][elem 16] bf16 ----------
// CDNA5 16-bit A 16x32 layout: lane L -> M = L&15, half = L>>4;
// elem e -> Klocal = 8*half + (e&7) + (e>=8 ? 16 : 0)
__global__ void k_packA(const float* __restrict__ X, const float* __restrict__ Y,
                        unsigned short* __restrict__ P, int rows, int Kdim, int nKt, int hasY) {
    int total = rows * Kdim;
    int i = blockIdx.x * blockDim.x + threadIdx.x;
    if (i >= total) return;
    int tile   = i >> 9;        // / 512
    int within = i & 511;
    int lane   = within >> 4;
    int e      = within & 15;
    int rt = tile / nKt;
    int kt = tile - rt * nKt;
    int half = lane >> 4;
    int M    = lane & 15;
    int Klocal = 8 * half + (e & 7) + ((e >= 8) ? 16 : 0);
    int row = rt * 16 + M;
    int K   = kt * 32 + Klocal;
    size_t idx = (size_t)row * Kdim + K;
    float v = X[idx];
    if (hasY) v += Y[idx];
    P[i] = f2bf(v);
}

// ---------- B-fragment pack: [ktile][ntile][lane 32][elem 16] bf16 ----------
// CDNA5 16-bit B 32x16 layout: lane L -> N = L&15, half = L>>4; elem e -> Klocal = 8*half + (e&7) + (e>=8 ? 16 : 0)
__global__ void k_packB(const float* __restrict__ W, unsigned short* __restrict__ P,
                        int Kdim, int Ndim) {
    int nNt = Ndim >> 4;
    int total = Kdim * Ndim;
    int i = blockIdx.x * blockDim.x + threadIdx.x;
    if (i >= total) return;
    int tile   = i >> 9;
    int within = i & 511;
    int lane   = within >> 4;
    int e      = within & 15;
    int kt = tile / nNt;
    int nt = tile - kt * nNt;
    int half = lane >> 4;
    int n    = lane & 15;
    int K = kt * 32 + 8 * half + (e & 7) + ((e >= 8) ? 16 : 0);
    int N = nt * 16 + n;
    P[i] = f2bf(W[K * Ndim + N]);
}

// ---------- WMMA GEMM: one wave per 16x16 output tile ----------
__global__ void k_gemm(const unsigned short* __restrict__ Apack,
                       const unsigned short* __restrict__ Bpack,
                       const float* __restrict__ bias,
                       float* __restrict__ Out,
                       int nRt, int nKt, int nNt, int Ndim) {
    int lane = threadIdx.x & 31;
    int wave = blockIdx.x * (blockDim.x >> 5) + (threadIdx.x >> 5);
    if (wave >= nRt * nNt) return;          // wave-uniform: EXEC stays all-ones
    int rt = wave / nNt;
    int nt = wave - rt * nNt;

    v8f acc = {0.f, 0.f, 0.f, 0.f, 0.f, 0.f, 0.f, 0.f};
    const v16bf* Abase = (const v16bf*)Apack + (size_t)rt * nKt * 32 + lane;
    const v16bf* Bbase = (const v16bf*)Bpack + (size_t)nt * 32 + lane;
    for (int kt = 0; kt < nKt; ++kt) {
        v16bf a = Abase[(size_t)kt * 32];
        v16bf b = Bbase[(size_t)kt * nNt * 32];
        acc = __builtin_amdgcn_wmma_f32_16x16x32_bf16(false, a, false, b,
                                                      (short)0, acc, false, false);
    }
    // C/D layout: VGPR r -> M = r + 8*(lane>>4), N = lane&15
    int half = lane >> 4;
    int n    = lane & 15;
    int col  = nt * 16 + n;
    float bv = bias ? bias[col] : 0.f;
    int rowbase = rt * 16 + 8 * half;
#pragma unroll
    for (int r = 0; r < 8; ++r)
        Out[(size_t)(rowbase + r) * Ndim + col] = acc[r] + bv;
}

// ---------- edge aggregation: agg[dst] += h[src], thread = (edge, float4 chunk) ----------
__global__ void k_agg(const int* __restrict__ src, const int* __restrict__ dst,
                      const float* __restrict__ H, float* __restrict__ Agg) {
    long long i = (long long)blockIdx.x * blockDim.x + threadIdx.x;
    const long long total = (long long)N_EDGES * 24;   // 96 floats = 24 float4
    if (i >= total) return;
    int e = (int)(i / 24);
    int g = (int)(i - (long long)e * 24);
    int s = src[e];
    int d = dst[e];
    float4 v = ((const float4*)(H + (size_t)s * HID))[g];
    float* out = Agg + (size_t)d * HID + g * 4;
    atomicAdd(out + 0, v.x);
    atomicAdd(out + 1, v.y);
    atomicAdd(out + 2, v.z);
    atomicAdd(out + 3, v.w);
}

// ---------- BatchNorm column stats (blockDim = 96, coalesced column reads) ----------
__global__ void k_bnstats(const float* __restrict__ Z, float* __restrict__ stats, int rows) {
    int c = threadIdx.x;                        // 0..95
    int rpb = (rows + gridDim.x - 1) / gridDim.x;
    int r0 = blockIdx.x * rpb;
    int r1 = r0 + rpb; if (r1 > rows) r1 = rows;
    float s = 0.f, q = 0.f;
    for (int r = r0; r < r1; ++r) {
        float v = Z[(size_t)r * HID + c];
        s += v; q += v * v;
    }
    atomicAdd(&stats[c], s);
    atomicAdd(&stats[HID + c], q);
}

__global__ void k_bnfin(float* __restrict__ stats, const float* __restrict__ gamma,
                        const float* __restrict__ beta) {
    int c = threadIdx.x;
    if (c >= HID) return;
    const float invN = 1.0f / (float)N_NODES;
    float mu  = stats[c] * invN;
    float var = stats[HID + c] * invN - mu * mu;
    float sc  = gamma[c] * rsqrtf(var + BN_EPS);
    stats[2 * HID + c] = sc;
    stats[3 * HID + c] = beta[c] - mu * sc;
}

__global__ void k_bnrelu(const float* __restrict__ Z, const float* __restrict__ stats,
                         float* __restrict__ H, int n) {
    int i = blockIdx.x * blockDim.x + threadIdx.x;
    if (i >= n) return;
    int c = i % HID;
    float v = Z[i] * stats[2 * HID + c] + stats[3 * HID + c];
    H[i] = v > 0.f ? v : 0.f;
}

// ---------- row log-softmax over 16 classes ----------
__global__ void k_logsoftmax(const float* __restrict__ L, float* __restrict__ Out, int rows) {
    int r = blockIdx.x * blockDim.x + threadIdx.x;
    if (r >= rows) return;
    const float* x = L + (size_t)r * NCLS;
    float m = x[0];
#pragma unroll
    for (int i = 1; i < NCLS; ++i) m = fmaxf(m, x[i]);
    float s = 0.f;
#pragma unroll
    for (int i = 0; i < NCLS; ++i) s += expf(x[i] - m);
    float lse = m + logf(s);
#pragma unroll
    for (int i = 0; i < NCLS; ++i) Out[(size_t)r * NCLS + i] = x[i] - lse;
}

// ---------- driver ----------
extern "C" void kernel_launch(void* const* d_in, const int* in_sizes, int n_in,
                              void* d_out, int out_size, void* d_ws, size_t ws_size,
                              hipStream_t stream) {
    const float* h    = (const float*)d_in[0];
    const int*   src  = (const int*)d_in[1];
    const int*   dst  = (const int*)d_in[2];
    const float* Wemb = (const float*)d_in[3];
    const float* bemb = (const float*)d_in[4];
    const float* W1   = (const float*)d_in[5];
    const float* b1   = (const float*)d_in[6];
    const float* g1   = (const float*)d_in[7];
    const float* be1  = (const float*)d_in[8];
    const float* W2   = (const float*)d_in[9];
    const float* b2   = (const float*)d_in[10];
    const float* g2   = (const float*)d_in[11];
    const float* be2  = (const float*)d_in[12];
    const float* Wr   = (const float*)d_in[13];
    const float* br   = (const float*)d_in[14];

    char* ws = (char*)d_ws;
    float*          A     = (float*)(ws + 0);            // 100000*96 f32 activations
    float*          B     = (float*)(ws + 38400000);     // 100000*96 f32 agg / z / logits
    unsigned short* P     = (unsigned short*)(ws + 76800000);   // 100000*128 bf16 A-frags
    unsigned short* PWemb = (unsigned short*)(ws + 102400000);
    unsigned short* PW1   = (unsigned short*)(ws + 102424576);
    unsigned short* PW2   = (unsigned short*)(ws + 102443008);
    unsigned short* PWr   = (unsigned short*)(ws + 102461440);
    float*          stats = (float*)(ws + 102464512);    // 384 f32

    const int nRT = N_NODES / 16;                        // 6250, exact
    const int NH  = N_NODES * HID;                       // 9,600,000

    // pack all weight matrices to B-fragment layout (deterministic, every call)
    k_packB<<<(IN_DIM * HID + 255) / 256, 256, 0, stream>>>(Wemb, PWemb, IN_DIM, HID);
    k_packB<<<(HID * HID + 255) / 256, 256, 0, stream>>>(W1, PW1, HID, HID);
    k_packB<<<(HID * HID + 255) / 256, 256, 0, stream>>>(W2, PW2, HID, HID);
    k_packB<<<(HID * NCLS + 255) / 256, 256, 0, stream>>>(Wr, PWr, HID, NCLS);

    // ---- embedding: A = h @ Wemb + bemb ----
    k_packA<<<(N_NODES * IN_DIM + 255) / 256, 256, 0, stream>>>(h, nullptr, P, N_NODES, IN_DIM, 4, 0);
    k_gemm<<<(nRT * 6 + 7) / 8, 256, 0, stream>>>(P, PWemb, bemb, A, nRT, 4, 6, HID);

    // ---- GIN layer 1 ----
    k_zero<<<37500, 256, 0, stream>>>(B, NH);
    k_agg<<<(int)(((long long)N_EDGES * 24 + 255) / 256), 256, 0, stream>>>(src, dst, A, B);
    k_packA<<<(NH + 255) / 256, 256, 0, stream>>>(A, B, P, N_NODES, HID, 3, 1);
    k_gemm<<<(nRT * 6 + 7) / 8, 256, 0, stream>>>(P, PW1, b1, B, nRT, 3, 6, HID);   // z1 -> B
    k_zero<<<2, 256, 0, stream>>>(stats, 384);
    k_bnstats<<<400, 96, 0, stream>>>(B, stats, N_NODES);
    k_bnfin<<<1, 96, 0, stream>>>(stats, g1, be1);
    k_bnrelu<<<(NH + 255) / 256, 256, 0, stream>>>(B, stats, A, NH);                // h1 -> A

    // ---- GIN layer 2 ----
    k_zero<<<37500, 256, 0, stream>>>(B, NH);
    k_agg<<<(int)(((long long)N_EDGES * 24 + 255) / 256), 256, 0, stream>>>(src, dst, A, B);
    k_packA<<<(NH + 255) / 256, 256, 0, stream>>>(A, B, P, N_NODES, HID, 3, 1);
    k_gemm<<<(nRT * 6 + 7) / 8, 256, 0, stream>>>(P, PW2, b2, B, nRT, 3, 6, HID);   // z2 -> B
    k_zero<<<2, 256, 0, stream>>>(stats, 384);
    k_bnstats<<<400, 96, 0, stream>>>(B, stats, N_NODES);
    k_bnfin<<<1, 96, 0, stream>>>(stats, g2, be2);
    k_bnrelu<<<(NH + 255) / 256, 256, 0, stream>>>(B, stats, A, NH);                // h2 -> A

    // ---- readout + log-softmax ----
    k_packA<<<(NH + 255) / 256, 256, 0, stream>>>(A, nullptr, P, N_NODES, HID, 3, 0);
    k_gemm<<<(nRT * 1 + 7) / 8, 256, 0, stream>>>(P, PWr, br, B, nRT, 3, 1, NCLS);  // logits -> B
    k_logsoftmax<<<(N_NODES + 255) / 256, 256, 0, stream>>>(B, (float*)d_out, N_NODES);
}
